// MyRNN_72541997629565
// MI455X (gfx1250) — matrix operand order repaired and checked by reading, not compiled, not run on hardware
//
#include <hip/hip_runtime.h>
#include <hip/hip_bf16.h>
#include <cstdint>
#include <cstddef>

// ---------------------------------------------------------------------------
// Problem constants (from reference): V=10000, E=100, T=80, B=1024, U=2048
// ---------------------------------------------------------------------------
#define RNN_B   1024
#define RNN_T   80
#define RNN_E   100
#define RNN_EP  128            // E padded up to multiple of BK
#define RNN_U   2048
#define KTOT0   (RNN_EP + RNN_U)   // 2176  (layer0: [x_pad | h0])
#define KTOT1   (2 * RNN_U)        // 4096  (layer1: [h0_new | h1])

// GEMM tiling
#define BM 128
#define BN 128
#define BK 64                  // two K=32 WMMA steps per LDS stage
#define LDSR 72                // LDS row stride in halfs (144 B = 36 banks, conflict-free)

typedef __attribute__((ext_vector_type(16))) __bf16 v16bf;
typedef __attribute__((ext_vector_type(8)))  float  v8f;

// ---------------------------------------------------------------------------
// CDNA5 async global->LDS path (ASYNCcnt-tracked), guarded for toolchain.
// Signature (from clang diagnostic): arg0 = int __vector(4) addrspace(1)*,
// arg1 = LDS pointer, arg2/arg3 = imm offset / cpol.
// ---------------------------------------------------------------------------
#if __has_builtin(__builtin_amdgcn_global_load_async_to_lds_b128)
#define HAVE_ASYNC_LDS 1
typedef int v4i_vs __attribute__((vector_size(16)));
typedef __attribute__((address_space(1))) v4i_vs* async_gptr_t;
typedef __attribute__((address_space(3))) v4i_vs* async_lptr_t;

__device__ __forceinline__ void async_cp16(const void* g, void* l) {
    __builtin_amdgcn_global_load_async_to_lds_b128(
        (async_gptr_t)g,
        (async_lptr_t)l,
        /*imm offset*/0, /*cpol*/0);
}
__device__ __forceinline__ void wait_async0() {
#if __has_builtin(__builtin_amdgcn_s_wait_asynccnt)
    __builtin_amdgcn_s_wait_asynccnt(0);
#else
    asm volatile("s_wait_asynccnt 0x0" ::: "memory");
#endif
}
#else
#define HAVE_ASYNC_LDS 0
#endif

// ---------------------------------------------------------------------------
// Hardware tanh (gfx1250 TRANS32 op) — avoids the branchy libm expansion.
// ---------------------------------------------------------------------------
__device__ __forceinline__ float fast_tanh(float x) {
#if __has_builtin(__builtin_amdgcn_tanhf)
    return __builtin_amdgcn_tanhf(x);
#else
    float r;
    asm volatile("v_tanh_f32 %0, %1\n\tv_nop" : "=v"(r) : "v"(x));
    return r;
#endif
}

// ---------------------------------------------------------------------------
// Zero a bf16 buffer
// ---------------------------------------------------------------------------
__global__ void zero_bf16(__bf16* __restrict__ p, int n) {
    int i = blockIdx.x * 256 + threadIdx.x;
    if (i < n) p[i] = (__bf16)0.0f;
}

// ---------------------------------------------------------------------------
// Embedding gather: xe[t][b][k] = (k < E) ? bf16(emb[tokens[b][t]][k]) : 0
// grid = T*B blocks of 128 threads
// ---------------------------------------------------------------------------
__global__ void gather_embed(const int* __restrict__ tokens,
                             const float* __restrict__ emb,
                             __bf16* __restrict__ xe) {
    int tb = blockIdx.x;            // t*B + b
    int t  = tb >> 10;              // / 1024
    int b  = tb & 1023;
    int k  = threadIdx.x;           // 0..127
    int tok = tokens[b * RNN_T + t];
    float v = (k < RNN_E) ? emb[(size_t)tok * RNN_E + k] : 0.0f;
    xe[((size_t)t * RNN_B + b) * RNN_EP + k] = (__bf16)v;
}

// ---------------------------------------------------------------------------
// Pack layer0 weights, transposed to [N=2048][K=2176] bf16:
//   k <  128 : Wx0[k][n] (zero-padded past E=100) ; k >= 128 : Wh0[k-128][n]
// ---------------------------------------------------------------------------
__global__ void pack_w0(const float* __restrict__ Wx0,
                        const float* __restrict__ Wh0,
                        __bf16* __restrict__ W0p) {
    int id = blockIdx.x * 256 + threadIdx.x;
    if (id >= RNN_U * KTOT0) return;
    int n = id / KTOT0;
    int k = id % KTOT0;
    float v;
    if (k < RNN_EP)
        v = (k < RNN_E) ? Wx0[(size_t)k * RNN_U + n] : 0.0f;
    else
        v = Wh0[(size_t)(k - RNN_EP) * RNN_U + n];
    W0p[id] = (__bf16)v;
}

// ---------------------------------------------------------------------------
// Pack layer1 weights, transposed to [N=2048][K=4096] bf16:
//   k < 2048 : Wx1[k][n] ; else Wh1[k-2048][n]
// ---------------------------------------------------------------------------
__global__ void pack_w1(const float* __restrict__ Wx1,
                        const float* __restrict__ Wh1,
                        __bf16* __restrict__ W1p) {
    int id = blockIdx.x * 256 + threadIdx.x;
    if (id >= RNN_U * KTOT1) return;
    int n = id / KTOT1;
    int k = id % KTOT1;
    float v = (k < RNN_U) ? Wx1[(size_t)k * RNN_U + n]
                          : Wh1[(size_t)(k - RNN_U) * RNN_U + n];
    W1p[id] = (__bf16)v;
}

// ---------------------------------------------------------------------------
// Fused GEMM + bias + tanh:
//   Hout[m][n] = tanh( sum_k A[m][k] * W[n][k] + bias[n] )
// A's K dim is the concat of region0 (A0, K0 cols, stride lda0) and region1
// (A1, K1 cols, stride lda1); K0, K1 are multiples of BK. W packed bf16 [N][K].
// Block: 256 threads = 8 wave32s; block tile 128x128; wave tile 64x32;
// 16 WMMAs per BK=64 LDS stage.
//  - HAVE_ASYNC_LDS: LDS double-buffer filled by GLOBAL_LOAD_ASYNC_TO_LDS_B128
//    (ASYNCcnt), one barrier per stage, no VGPR staging.
//  - fallback: register double-buffer with named int4 stage registers.
// ---------------------------------------------------------------------------

// compute global source pointers for stage k0_
#define GSRC(k0_)                                                             \
    const __bf16* src_;                                                       \
    int stride_, kc_;                                                         \
    if ((k0_) < K0) { src_ = A0; stride_ = lda0; kc_ = (k0_);      }          \
    else            { src_ = A1; stride_ = lda1; kc_ = (k0_) - K0; }          \
    const __bf16* ga_ = src_ + (size_t)(blockM + ldRow) * stride_ + kc_ + ldCol; \
    const __bf16* gb_ = Wp   + (size_t)(blockN + ldRow) * ldw     + (k0_) + ldCol;

__global__ __launch_bounds__(256)
void rnn_gemm_tanh(const __bf16* __restrict__ A0, int lda0, int K0,
                   const __bf16* __restrict__ A1, int lda1, int K1,
                   const __bf16* __restrict__ Wp, int ldw,
                   const float* __restrict__ bias,
                   __bf16* __restrict__ Hout) {
    const int tid  = threadIdx.x;
    const int lane = tid & 31;
    const int wave = tid >> 5;
    const int sub  = lane >> 4;        // 0 or 1 (half-wave)
    const int l16  = lane & 15;

    const int wm = (wave >> 2) * 64;   // wave M offset in block tile (0 or 64)
    const int wn = (wave & 3) * 32;    // wave N offset in block tile (0..96)

    const int blockN = blockIdx.x * BN;
    const int blockM = blockIdx.y * BM;

    // loader mapping: row = tid>>1 (0..127), 32-half column group per thread
    const int ldRow = tid >> 1;
    const int ldCol = (tid & 1) * 32;  // halfs

    v8f c[4][2];
#pragma unroll
    for (int mi = 0; mi < 4; ++mi)
#pragma unroll
        for (int ni = 0; ni < 2; ++ni)
            c[mi][ni] = (v8f)(0.0f);

    const int Ktot = K0 + K1;

#if HAVE_ASYNC_LDS
    // ======================= async double-buffer path =======================
    __shared__ __bf16 As[2][BM * LDSR];
    __shared__ __bf16 Bs[2][BN * LDSR];

#define ASTAGE(k0_, buf_)                                                     \
    do {                                                                      \
        GSRC(k0_);                                                            \
        __bf16* la_ = &As[buf_][(size_t)ldRow * LDSR + ldCol];                \
        __bf16* lb_ = &Bs[buf_][(size_t)ldRow * LDSR + ldCol];                \
        async_cp16(ga_,      la_);      async_cp16(ga_ + 8,  la_ + 8);        \
        async_cp16(ga_ + 16, la_ + 16); async_cp16(ga_ + 24, la_ + 24);       \
        async_cp16(gb_,      lb_);      async_cp16(gb_ + 8,  lb_ + 8);        \
        async_cp16(gb_ + 16, lb_ + 16); async_cp16(gb_ + 24, lb_ + 24);       \
    } while (0)

    ASTAGE(0, 0);
    int cur = 0;

    for (int k0 = 0; k0 < Ktot; k0 += BK) {
        wait_async0();          // this wave's copies into As/Bs[cur] done
        __syncthreads();        // ... and everyone else's

        if (k0 + BK < Ktot) {
            ASTAGE(k0 + BK, cur ^ 1);   // stream next stage while we compute
            if (k0 + 2 * BK < Ktot)
                __builtin_prefetch(Wp + (size_t)(blockN + ldRow) * ldw +
                                   (k0 + 2 * BK) + ldCol, 0, 0);
        }

        const __bf16* Ab = As[cur];
        const __bf16* Bb = Bs[cur];
#else
    // ==================== register double-buffer fallback ===================
    __shared__ __bf16 As[BM * LDSR];
    __shared__ __bf16 Bs[BN * LDSR];

    int4 pa0, pa1, pa2, pa3;
    int4 pb0, pb1, pb2, pb3;

#define GLOAD(k0_)                                                            \
    do {                                                                      \
        GSRC(k0_);                                                            \
        pa0 = *(const int4*)(ga_);      pa1 = *(const int4*)(ga_ + 8);        \
        pa2 = *(const int4*)(ga_ + 16); pa3 = *(const int4*)(ga_ + 24);       \
        pb0 = *(const int4*)(gb_);      pb1 = *(const int4*)(gb_ + 8);        \
        pb2 = *(const int4*)(gb_ + 16); pb3 = *(const int4*)(gb_ + 24);       \
    } while (0)

    GLOAD(0);

    __bf16* sa = &As[(size_t)ldRow * LDSR + ldCol];
    __bf16* sb = &Bs[(size_t)ldRow * LDSR + ldCol];

    for (int k0 = 0; k0 < Ktot; k0 += BK) {
        *(int4*)(sa)      = pa0;
        *(int4*)(sa + 8)  = pa1;
        *(int4*)(sa + 16) = pa2;
        *(int4*)(sa + 24) = pa3;
        *(int4*)(sb)      = pb0;
        *(int4*)(sb + 8)  = pb1;
        *(int4*)(sb + 16) = pb2;
        *(int4*)(sb + 24) = pb3;
        __syncthreads();

        if (k0 + BK < Ktot) {
            GLOAD(k0 + BK);
            if (k0 + 2 * BK < Ktot)
                __builtin_prefetch(Wp + (size_t)(blockN + ldRow) * ldw +
                                   (k0 + 2 * BK) + ldCol, 0, 0);
        }

        const __bf16* Ab = As;
        const __bf16* Bb = Bs;
#endif
        // ---- 16 WMMAs from the LDS stage (2 K-steps of 32) ----
#pragma unroll
        for (int ks = 0; ks < BK; ks += 32) {
            union Frag { v16bf v; int4 q[2]; };
            Frag af0, af1, af2, af3, bf0, bf1;

            // A frags: lane<16 -> K {0..7, 16..23}; lane>=16 -> K {8..15, 24..31}
            const __bf16* ap0 = &Ab[(size_t)(wm +  0 + l16) * LDSR + ks];
            const __bf16* ap1 = &Ab[(size_t)(wm + 16 + l16) * LDSR + ks];
            const __bf16* ap2 = &Ab[(size_t)(wm + 32 + l16) * LDSR + ks];
            const __bf16* ap3 = &Ab[(size_t)(wm + 48 + l16) * LDSR + ks];
            af0.q[0] = *(const int4*)(ap0 + sub * 8);
            af0.q[1] = *(const int4*)(ap0 + 16 + sub * 8);
            af1.q[0] = *(const int4*)(ap1 + sub * 8);
            af1.q[1] = *(const int4*)(ap1 + 16 + sub * 8);
            af2.q[0] = *(const int4*)(ap2 + sub * 8);
            af2.q[1] = *(const int4*)(ap2 + 16 + sub * 8);
            af3.q[0] = *(const int4*)(ap3 + sub * 8);
            af3.q[1] = *(const int4*)(ap3 + 16 + sub * 8);

            // B frags: lane<16 -> K {0..7, 16..23}; lane>=16 -> K {8..15, 24..31}
            const __bf16* bp0 = &Bb[(size_t)(wn +  0 + l16) * LDSR + ks + sub * 8];
            const __bf16* bp1 = &Bb[(size_t)(wn + 16 + l16) * LDSR + ks + sub * 8];
            bf0.q[0] = *(const int4*)(bp0);
            bf0.q[1] = *(const int4*)(bp0 + 16);
            bf1.q[0] = *(const int4*)(bp1);
            bf1.q[1] = *(const int4*)(bp1 + 16);

            c[0][0] = __builtin_amdgcn_wmma_f32_16x16x32_bf16(false, af0.v, false, bf0.v, (short)0, c[0][0], false, false);
            c[0][1] = __builtin_amdgcn_wmma_f32_16x16x32_bf16(false, af0.v, false, bf1.v, (short)0, c[0][1], false, false);
            c[1][0] = __builtin_amdgcn_wmma_f32_16x16x32_bf16(false, af1.v, false, bf0.v, (short)0, c[1][0], false, false);
            c[1][1] = __builtin_amdgcn_wmma_f32_16x16x32_bf16(false, af1.v, false, bf1.v, (short)0, c[1][1], false, false);
            c[2][0] = __builtin_amdgcn_wmma_f32_16x16x32_bf16(false, af2.v, false, bf0.v, (short)0, c[2][0], false, false);
            c[2][1] = __builtin_amdgcn_wmma_f32_16x16x32_bf16(false, af2.v, false, bf1.v, (short)0, c[2][1], false, false);
            c[3][0] = __builtin_amdgcn_wmma_f32_16x16x32_bf16(false, af3.v, false, bf0.v, (short)0, c[3][0], false, false);
            c[3][1] = __builtin_amdgcn_wmma_f32_16x16x32_bf16(false, af3.v, false, bf1.v, (short)0, c[3][1], false, false);
        }

#if HAVE_ASYNC_LDS
        cur ^= 1;               // wave reads of As/Bs[cur] drained before its
                                // next barrier; no tail barrier needed
    }
#else
        __syncthreads();
    }
#endif

    // --------- epilogue: bias + hardware tanh + bf16 store ----------
#pragma unroll
    for (int ni = 0; ni < 2; ++ni) {
        const int n = blockN + wn + ni * 16 + l16;   // C layout: N = lane&15
        const float bv = bias[n];
#pragma unroll
        for (int mi = 0; mi < 4; ++mi) {
#pragma unroll
            for (int i = 0; i < 8; ++i) {
                const int m = blockM + wm + mi * 16 + sub * 8 + i;  // M = i + 8*sub
                float v = fast_tanh(c[mi][ni][i] + bv);
                Hout[(size_t)m * RNN_U + n] = (__bf16)v;
            }
        }
    }
}

// ---------------------------------------------------------------------------
// Output head: out[b] = sigmoid( dot(h1[b,:], Wo) + bo )
// one wave32 per batch row; 8 rows per 256-thread block; grid = 128
// ---------------------------------------------------------------------------
__global__ __launch_bounds__(256)
void final_head(const __bf16* __restrict__ h1,
                const float* __restrict__ Wo,
                const float* __restrict__ bo,
                float* __restrict__ out) {
    const int wave = threadIdx.x >> 5;
    const int lane = threadIdx.x & 31;
    const int b = blockIdx.x * 8 + wave;
    const __bf16* row = h1 + (size_t)b * RNN_U;
    float s = 0.0f;
    for (int u = lane; u < RNN_U; u += 32)
        s += (float)row[u] * Wo[u];
#pragma unroll
    for (int off = 16; off > 0; off >>= 1)
        s += __shfl_down(s, off, 32);
    if (lane == 0)
        out[b] = 1.0f / (1.0f + __expf(-(s + bo[0])));
}

// ---------------------------------------------------------------------------
// Host-side orchestration
// Inputs (setup_inputs order):
//  0 tokens [B,T] i32 | 1 emb [V,E] f32 | 2 Wx0 [E,U] | 3 Wh0 [U,U] | 4 b0 [U]
//  5 Wx1 [U,U] | 6 Wh1 [U,U] | 7 b1 [U] | 8 Wo [U,1] | 9 bo [1]
// Output: [B,1] f32 sigmoid
// ---------------------------------------------------------------------------
extern "C" void kernel_launch(void* const* d_in, const int* in_sizes, int n_in,
                              void* d_out, int out_size, void* d_ws, size_t ws_size,
                              hipStream_t stream) {
    (void)in_sizes; (void)n_in; (void)out_size; (void)ws_size;

    const int*   tokens = (const int*)  d_in[0];
    const float* emb    = (const float*)d_in[1];
    const float* Wx0    = (const float*)d_in[2];
    const float* Wh0    = (const float*)d_in[3];
    const float* b0     = (const float*)d_in[4];
    const float* Wx1    = (const float*)d_in[5];
    const float* Wh1    = (const float*)d_in[6];
    const float* b1     = (const float*)d_in[7];
    const float* Wo     = (const float*)d_in[8];
    const float* bo     = (const float*)d_in[9];
    float*       out    = (float*)d_out;

    // ---- carve workspace (bf16 buffers), 256B aligned ----
    char* wp = (char*)d_ws;
    auto carve = [&](size_t bytes) -> void* {
        void* r = (void*)wp;
        wp += (bytes + 255) & ~(size_t)255;
        return r;
    };
    __bf16* xe  = (__bf16*)carve((size_t)RNN_T * RNN_B * RNN_EP * 2);  // ~21 MB
    __bf16* W0p = (__bf16*)carve((size_t)RNN_U * KTOT0 * 2);           // ~8.9 MB
    __bf16* W1p = (__bf16*)carve((size_t)RNN_U * KTOT1 * 2);           // ~16.8 MB
    __bf16* h0a = (__bf16*)carve((size_t)RNN_B * RNN_U * 2);           // 4 MB each
    __bf16* h0b = (__bf16*)carve((size_t)RNN_B * RNN_U * 2);
    __bf16* h1a = (__bf16*)carve((size_t)RNN_B * RNN_U * 2);
    __bf16* h1b = (__bf16*)carve((size_t)RNN_B * RNN_U * 2);

    // ---- prep: zero initial states, gather embeddings, pack weights ----
    {
        const int nh = RNN_B * RNN_U;
        zero_bf16<<<(nh + 255) / 256, 256, 0, stream>>>(h0a, nh);
        zero_bf16<<<(nh + 255) / 256, 256, 0, stream>>>(h1a, nh);

        gather_embed<<<RNN_T * RNN_B, RNN_EP, 0, stream>>>(tokens, emb, xe);

        const int n0 = RNN_U * KTOT0;
        pack_w0<<<(n0 + 255) / 256, 256, 0, stream>>>(Wx0, Wh0, W0p);
        const int n1 = RNN_U * KTOT1;
        pack_w1<<<(n1 + 255) / 256, 256, 0, stream>>>(Wx1, Wh1, W1p);
    }

    // ---- recurrent loop: one launch per layer per step (grid-wide sync) ----
    dim3 grid(RNN_U / BN, RNN_B / BM);   // (16, 8)
    __bf16 *h0cur = h0a, *h0nxt = h0b, *h1cur = h1a, *h1nxt = h1b;
    for (int t = 0; t < RNN_T; ++t) {
        // layer0: A = [x_t(pad 128) | h0cur], W0p [2048][2176]
        rnn_gemm_tanh<<<grid, 256, 0, stream>>>(
            xe + (size_t)t * RNN_B * RNN_EP, RNN_EP, RNN_EP,
            h0cur, RNN_U, RNN_U,
            W0p, KTOT0, b0, h0nxt);
        // layer1: A = [h0nxt | h1cur], W1p [2048][4096]
        rnn_gemm_tanh<<<grid, 256, 0, stream>>>(
            h0nxt, RNN_U, RNN_U,
            h1cur, RNN_U, RNN_U,
            W1p, KTOT1, b1, h1nxt);
        // ping-pong
        __bf16* tmp;
        tmp = h0cur; h0cur = h0nxt; h0nxt = tmp;
        tmp = h1cur; h1cur = h1nxt; h1nxt = tmp;
    }

    // ---- head: logits + sigmoid ----
    final_head<<<RNN_B / 8, 256, 0, stream>>>(h1cur, Wo, bo, out);
}
